// RelationMessagePassing_84112639525248
// MI455X (gfx1250) — compile-verified, matrix operand order checked by reading, not run on hardware
//
#include <hip/hip_runtime.h>
#include <hip/hip_bf16.h>

// ---------------------------------------------------------------------------
// Types for CDNA5 WMMA (wave32): D[16x16,f32] = A[16x32,bf16] x B[32x16,bf16] + C
// ---------------------------------------------------------------------------
typedef __attribute__((ext_vector_type(16))) __bf16 bf16x16;
typedef __attribute__((ext_vector_type(8)))  __bf16 bf16x8;
typedef __attribute__((ext_vector_type(4)))  __bf16 bf16x4;
typedef __attribute__((ext_vector_type(8)))  float  f32x8;

__device__ __forceinline__ f32x8 wmma_bf16(bf16x16 a, bf16x16 b, f32x8 c) {
  return __builtin_amdgcn_wmma_f32_16x16x32_bf16(
      /*neg_a=*/false, a, /*neg_b=*/false, b,
      /*c_mod=*/(short)0, c, /*reuse_a=*/false, /*reuse_b=*/false);
}

__device__ __forceinline__ f32x8 zero8() {
  f32x8 c;
#pragma unroll
  for (int r = 0; r < 8; ++r) c[r] = 0.f;
  return c;
}

// A-fragment from bf16 LDS: two contiguous 16B loads per lane, no conversion.
__device__ __forceinline__ bf16x16 load_a_bf16(const __bf16* X, int lda, int kbase) {
  const int lane = threadIdx.x & 31;
  const int i = lane & 15, h8 = (lane >> 4) << 3;
  const __bf16* row = X + i * lda + kbase + h8;
  bf16x8 lo = *(const bf16x8*)row;
  bf16x8 hi = *(const bf16x8*)(row + 16);
  return __builtin_shufflevector(lo, hi, 0, 1, 2, 3, 4, 5, 6, 7,
                                 8, 9, 10, 11, 12, 13, 14, 15);
}

// B-fragment from pre-swizzled bf16 weights: one contiguous 32B chunk per lane.
__device__ __forceinline__ bf16x16 load_b_swz(const __bf16* Wswz, int bCT, int kt, int nt) {
  const int lane = threadIdx.x & 31;
  const __bf16* p = Wswz + ((((kt * bCT + nt) << 5) + lane) << 4);
  return *(const bf16x16*)p;
}

// C layout: element r -> row (r + 8*half), col (lane&15)
__device__ __forceinline__ void store_c_bf16(__bf16* X, int ld, int nbase, f32x8 c, bool relu) {
  const int lane = threadIdx.x & 31;
  const int i = lane & 15, half = lane >> 4;
#pragma unroll
  for (int r = 0; r < 8; ++r) {
    float v = c[r];
    if (relu) v = fmaxf(v, 0.f);
    X[(r + 8 * half) * ld + nbase + i] = (__bf16)v;
  }
}

__device__ __forceinline__ f32x8 bias_acc(const float* b, int nbase) {
  const float bv = b[nbase + (threadIdx.x & 15)];
  f32x8 c;
#pragma unroll
  for (int r = 0; r < 8; ++r) c[r] = bv;
  return c;
}

// Multi-row-tile GEMM: B fragment loaded once, reused across RT row tiles.
template <int RT>
__device__ __forceinline__ void gemm_multi(const __bf16* A, int lda, int K,
                                           const __bf16* Bswz, int bCT, int nt, f32x8* acc) {
#pragma unroll
  for (int kt = 0; kt < K / 32; ++kt) {
    bf16x16 bf = load_b_swz(Bswz, bCT, kt, nt);
#pragma unroll
    for (int rt = 0; rt < RT; ++rt) {
      bf16x16 af = load_a_bf16(A + rt * 16 * lda, lda, kt * 32);
      acc[rt] = wmma_bf16(af, bf, acc[rt]);
    }
  }
}

// ---------------------------------------------------------------------------
// Prep kernels
// ---------------------------------------------------------------------------
__global__ void zero_kernel(float* __restrict__ p, int n) {
  int i = blockIdx.x * 256 + threadIdx.x;
  if (i < n) p[i] = 0.f;
}

// Swizzle f32 weight [K][Ncols] -> bf16 frag-major: dst[((kt*CT+nt)*32+lane)*16+e]
__global__ void swz_kernel(const float* __restrict__ W, __bf16* __restrict__ dst,
                           int K, int Ncols) {
  int i = blockIdx.x * 256 + threadIdx.x;
  if (i >= K * Ncols) return;
  const int e = i & 15;
  const int lane = (i >> 4) & 31;
  const int frag = i >> 9;
  const int CT = Ncols >> 4;
  const int kt = frag / CT, nt = frag - kt * CT;
  const int li = lane & 15, h8 = (lane >> 4) << 3;
  const int k = kt * 32 + ((e < 8) ? (h8 + e) : (16 + h8 + (e - 8)));
  const int n = nt * 16 + li;
  dst[i] = (__bf16)W[k * Ncols + n];
}

__global__ void hist_kernel(const int* __restrict__ idx, int n, float* __restrict__ cnt) {
  int i = blockIdx.x * 256 + threadIdx.x;
  if (i < n) atomicAdd(&cnt[idx[i]], 1.0f);
}

__device__ __forceinline__ bf16x4 cvt4(float4 v) {
  bf16x4 b;
  b[0] = (__bf16)v.x; b[1] = (__bf16)v.y; b[2] = (__bf16)v.z; b[3] = (__bf16)v.w;
  return b;
}

// ---------------------------------------------------------------------------
// q = node_states @ Wq   (32 rows per block; wave = col tile, 2 row tiles)
// ---------------------------------------------------------------------------
__launch_bounds__(128)
__global__ void q_kernel(const float* __restrict__ ns, const __bf16* __restrict__ Wq,
                         float* __restrict__ q, int N) {
  __shared__ __bf16 xs[32 * 72];
  const int tid = threadIdx.x, wave = tid >> 5, tile = blockIdx.x;
  const int row0 = tile * 32;
  const bool full = (row0 + 32) <= N;
  for (int i = tid; i < 32 * 16; i += 128) {  // float4 granularity
    const int row = i >> 4, h4 = i & 15;
    const int r = row0 + row;
    float4 v = (full || r < N) ? ((const float4*)ns)[r * 16 + h4]
                               : make_float4(0.f, 0.f, 0.f, 0.f);
    *(bf16x4*)(xs + row * 72 + h4 * 4) = cvt4(v);
  }
  __syncthreads();
  f32x8 acc[2] = {zero8(), zero8()};
  gemm_multi<2>(xs, 72, 64, Wq, 4, wave, acc);
  const int lane = tid & 31, i = lane & 15, half = lane >> 4;
#pragma unroll
  for (int rt = 0; rt < 2; ++rt) {
    float* qp = q + (size_t)(row0 + rt * 16) * 64 + wave * 16 + i;
    if (full) {
#pragma unroll
      for (int r = 0; r < 8; ++r) qp[(r + 8 * half) * 64] = acc[rt][r];
    } else {
#pragma unroll
      for (int r = 0; r < 8; ++r)
        if (row0 + rt * 16 + r + 8 * half < N) qp[(r + 8 * half) * 64] = acc[rt][r];
    }
  }
}

// ---------------------------------------------------------------------------
// Fused relation kernel: gather -> MLP1(relu) -> MLP2 -> K/V proj -> scores ->
// exp -> atomic softmax accumulation. 16*RT tuples per block, 4 waves.
// ---------------------------------------------------------------------------
template <int ARITY, int RT>
__launch_bounds__(128)
__global__ void rel_kernel(const float* __restrict__ ns, const int* __restrict__ idx, int T,
                           const __bf16* __restrict__ W1, const float* __restrict__ b1,
                           const __bf16* __restrict__ W2, const float* __restrict__ b2,
                           const __bf16* __restrict__ Wk, const __bf16* __restrict__ Wv,
                           const float* __restrict__ q, const float* __restrict__ cnt,
                           float* __restrict__ den, float* __restrict__ num) {
  constexpr int D = ARITY * 64;
  constexpr int LDX = D + 8;             // bf16 pad: conflict-free, keeps 16B row align
  constexpr int TROWS = 16 * RT;         // tuples per block
  constexpr int ROWS = TROWS * ARITY;    // message rows per block
  constexpr int MT = ARITY * RT;         // 16-row message tiles
  __shared__ __bf16 xs[TROWS * LDX];
  __shared__ __bf16 h1s[TROWS * LDX];
  __shared__ __bf16 ms[ROWS * 72];       // messages (padded rows)
  __shared__ __bf16 ks[ROWS * 64];
  __shared__ __bf16 vs[ROWS * 64];

  const int tid = threadIdx.x, wave = tid >> 5, lane = tid & 31;
  const int tup0 = blockIdx.x * TROWS;

  // 1) gather x = concat(node_states[idx[t,0..A-1]]) into bf16 LDS
  constexpr int D4 = D / 4;
  for (int i = tid; i < TROWS * D4; i += 128) {
    const int row = i / D4, c4 = i - row * D4;
    const int pos = c4 >> 4, h4 = c4 & 15;
    const int tup = tup0 + row;
    const int node = (tup < T) ? idx[tup * ARITY + pos] : 0;
    float4 v = ((const float4*)ns)[node * 16 + h4];
    *(bf16x4*)(xs + row * LDX + c4 * 4) = cvt4(v);
  }
  __syncthreads();

  // 2) h1 = relu(x @ W1 + b1)
  constexpr int CT = D / 16;
  for (int ct = wave; ct < CT; ct += 4) {
    f32x8 acc[RT];
#pragma unroll
    for (int rt = 0; rt < RT; ++rt) acc[rt] = bias_acc(b1, ct * 16);
    gemm_multi<RT>(xs, LDX, D, W1, CT, ct, acc);
#pragma unroll
    for (int rt = 0; rt < RT; ++rt)
      store_c_bf16(h1s + rt * 16 * LDX, LDX, ct * 16, acc[rt], true);
  }
  __syncthreads();

  // 3) m = h1 @ W2 + b2, stored re-shaped as message rows [ROWS][64] bf16
  for (int ct = wave; ct < CT; ct += 4) {
    f32x8 acc[RT];
#pragma unroll
    for (int rt = 0; rt < RT; ++rt) acc[rt] = bias_acc(b2, ct * 16);
    gemm_multi<RT>(h1s, LDX, D, W2, CT, ct, acc);
    const int i = lane & 15, half = lane >> 4;
    const int col = ct * 16 + i;
    const int pos = col >> 6, mc = col & 63;
#pragma unroll
    for (int rt = 0; rt < RT; ++rt)
#pragma unroll
      for (int r = 0; r < 8; ++r) {
        const int trow = rt * 16 + r + 8 * half;
        ms[(trow * ARITY + pos) * 72 + mc] = (__bf16)acc[rt][r];
      }
  }
  __syncthreads();

  // 4) k = m @ Wk ; v = m @ Wv. One col tile per wave; Wk/Wv fragments loaded
  //    once per block; each A fragment feeds both k and v WMMAs.
  {
    const int nt = wave;
    f32x8 ak[MT], av[MT];
#pragma unroll
    for (int mt = 0; mt < MT; ++mt) { ak[mt] = zero8(); av[mt] = zero8(); }
#pragma unroll
    for (int kt = 0; kt < 2; ++kt) {
      bf16x16 bk = load_b_swz(Wk, 4, kt, nt);
      bf16x16 bv = load_b_swz(Wv, 4, kt, nt);
#pragma unroll
      for (int mt = 0; mt < MT; ++mt) {
        bf16x16 af = load_a_bf16(ms + mt * 16 * 72, 72, kt * 32);
        ak[mt] = wmma_bf16(af, bk, ak[mt]);
        av[mt] = wmma_bf16(af, bv, av[mt]);
      }
    }
#pragma unroll
    for (int mt = 0; mt < MT; ++mt) {
      store_c_bf16(ks + mt * 16 * 64, 64, nt * 16, ak[mt], false);
      store_c_bf16(vs + mt * 16 * 64, 64, nt * 16, av[mt], false);
    }
  }
  __syncthreads();

  // 5) per-message-row: score = dot(q[seg],k)*rsqrt(count); e = exp(score);
  //    den[seg] += e; num[seg,:] += e * v   (atomics, L2-resident arrays)
  for (int j = wave; j < ROWS; j += 4) {
    const int tup = tup0 + j / ARITY;
    if (tup >= T) continue;
    const int seg = idx[tup * ARITY + (j % ARITY)];
    const __bf16* kr = ks + j * 64;
    const __bf16* vr = vs + j * 64;
    const float* qr = q + (size_t)seg * 64;
    float p = qr[lane] * (float)kr[lane] + qr[lane + 32] * (float)kr[lane + 32];
#pragma unroll
    for (int off = 16; off > 0; off >>= 1) p += __shfl_xor(p, off, 32);
    const float e = __expf(p * rsqrtf(fmaxf(cnt[seg], 1.0f)));
    if (lane == 0) atomicAdd(&den[seg], e);
    atomicAdd(&num[(size_t)seg * 64 + lane], e * (float)vr[lane]);
    atomicAdd(&num[(size_t)seg * 64 + lane + 32], e * (float)vr[lane + 32]);
  }
}

// ---------------------------------------------------------------------------
// Final update MLP: z = [num/den', ns]; out = relu(z@uW1+ub1)@uW2+ub2
// 32 rows per block; B fragments reused across 2 row tiles.
// ---------------------------------------------------------------------------
__launch_bounds__(128)
__global__ void out_kernel(const float* __restrict__ ns, const float* __restrict__ num,
                           const float* __restrict__ den,
                           const __bf16* __restrict__ uW1, const float* __restrict__ ub1,
                           const __bf16* __restrict__ uW2, const float* __restrict__ ub2,
                           float* __restrict__ out, int N) {
  __shared__ __bf16 zs[32 * 136];
  __shared__ __bf16 h1s[32 * 136];
  const int tid = threadIdx.x, wave = tid >> 5, tile = blockIdx.x;
  const int row0 = tile * 32;
  const bool full = (row0 + 32) <= N;

  for (int i = tid; i < 32 * 64; i += 128) {
    const int row = i >> 6, h = i & 63;
    const int r = row0 + row;
    float a = 0.f, x = 0.f;
    if (full || r < N) {
      const float d = den[r];
      a = num[(size_t)r * 64 + h] / (d > 0.f ? d : 1.f);
      x = ns[(size_t)r * 64 + h];
    }
    zs[row * 136 + h] = (__bf16)a;
    zs[row * 136 + 64 + h] = (__bf16)x;
  }
  __syncthreads();

  for (int ct = wave; ct < 8; ct += 4) {
    f32x8 acc[2];
#pragma unroll
    for (int rt = 0; rt < 2; ++rt) acc[rt] = bias_acc(ub1, ct * 16);
    gemm_multi<2>(zs, 136, 128, uW1, 8, ct, acc);
#pragma unroll
    for (int rt = 0; rt < 2; ++rt)
      store_c_bf16(h1s + rt * 16 * 136, 136, ct * 16, acc[rt], true);
  }
  __syncthreads();

  {
    f32x8 acc[2];
#pragma unroll
    for (int rt = 0; rt < 2; ++rt) acc[rt] = bias_acc(ub2, wave * 16);
    gemm_multi<2>(h1s, 136, 128, uW2, 4, wave, acc);
    const int lane = tid & 31, i = lane & 15, half = lane >> 4;
#pragma unroll
    for (int rt = 0; rt < 2; ++rt) {
      float* op = out + (size_t)(row0 + rt * 16) * 64 + wave * 16 + i;
      if (full) {
#pragma unroll
        for (int r = 0; r < 8; ++r) op[(r + 8 * half) * 64] = acc[rt][r];
      } else {
#pragma unroll
        for (int r = 0; r < 8; ++r)
          if (row0 + rt * 16 + r + 8 * half < N) op[(r + 8 * half) * 64] = acc[rt][r];
      }
    }
  }
}

// ---------------------------------------------------------------------------
// Host orchestration
// ---------------------------------------------------------------------------
extern "C" void kernel_launch(void* const* d_in, const int* in_sizes, int n_in,
                              void* d_out, int out_size, void* d_ws, size_t ws_size,
                              hipStream_t stream) {
  const float* ns   = (const float*)d_in[0];
  const int* idx0   = (const int*)d_in[1];
  const int* idx1   = (const int*)d_in[2];
  const int* idx2   = (const int*)d_in[3];
  const float* r0W1 = (const float*)d_in[4];  const float* r0b1 = (const float*)d_in[5];
  const float* r0W2 = (const float*)d_in[6];  const float* r0b2 = (const float*)d_in[7];
  const float* r1W1 = (const float*)d_in[8];  const float* r1b1 = (const float*)d_in[9];
  const float* r1W2 = (const float*)d_in[10]; const float* r1b2 = (const float*)d_in[11];
  const float* r2W1 = (const float*)d_in[12]; const float* r2b1 = (const float*)d_in[13];
  const float* r2W2 = (const float*)d_in[14]; const float* r2b2 = (const float*)d_in[15];
  const float* Wq   = (const float*)d_in[16];
  const float* Wk   = (const float*)d_in[17];
  const float* Wv   = (const float*)d_in[18];
  const float* uW1  = (const float*)d_in[19]; const float* ub1 = (const float*)d_in[20];
  const float* uW2  = (const float*)d_in[21]; const float* ub2 = (const float*)d_in[22];

  const int N  = in_sizes[0] / 64;
  const int L0 = in_sizes[1], L1 = in_sizes[2], L2 = in_sizes[3];
  const int T0 = L0, T1 = L1 / 2, T2 = L2 / 3;

  // workspace: q[N*64] | num[N*64] | den[N] | cnt[N] | bf16 swizzled weights
  float* qb  = (float*)d_ws;
  float* num = qb + (size_t)N * 64;
  float* den = num + (size_t)N * 64;
  float* cnt = den + N;
  __bf16* wb = (__bf16*)(cnt + N);
  size_t need = (size_t)N * 130 * 4 + 151552 * 2;
  if (ws_size < need) return;

  auto alloc_bf = [&](int n) { __bf16* p = wb; wb += n; return p; };
  __bf16* b_r0W1 = alloc_bf(64 * 64);   __bf16* b_r0W2 = alloc_bf(64 * 64);
  __bf16* b_r1W1 = alloc_bf(128 * 128); __bf16* b_r1W2 = alloc_bf(128 * 128);
  __bf16* b_r2W1 = alloc_bf(192 * 192); __bf16* b_r2W2 = alloc_bf(192 * 192);
  __bf16* b_Wq   = alloc_bf(64 * 64);   __bf16* b_Wk   = alloc_bf(64 * 64);
  __bf16* b_Wv   = alloc_bf(64 * 64);
  __bf16* b_uW1  = alloc_bf(128 * 128); __bf16* b_uW2  = alloc_bf(128 * 64);

  // 0) zero accumulators (num|den|cnt are contiguous: 66*N floats)
  const int zn = N * 66;
  zero_kernel<<<(zn + 255) / 256, 256, 0, stream>>>(num, zn);

  // 1) swizzle all weights to bf16 frag order
  auto swz = [&](const float* W, __bf16* dst, int K, int Nc) {
    swz_kernel<<<(K * Nc + 255) / 256, 256, 0, stream>>>(W, dst, K, Nc);
  };
  swz(r0W1, b_r0W1, 64, 64);    swz(r0W2, b_r0W2, 64, 64);
  swz(r1W1, b_r1W1, 128, 128);  swz(r1W2, b_r1W2, 128, 128);
  swz(r2W1, b_r2W1, 192, 192);  swz(r2W2, b_r2W2, 192, 192);
  swz(Wq, b_Wq, 64, 64);        swz(Wk, b_Wk, 64, 64);  swz(Wv, b_Wv, 64, 64);
  swz(uW1, b_uW1, 128, 128);    swz(uW2, b_uW2, 128, 64);

  // 2) segment counts
  hist_kernel<<<(L0 + 255) / 256, 256, 0, stream>>>(idx0, L0, cnt);
  hist_kernel<<<(L1 + 255) / 256, 256, 0, stream>>>(idx1, L1, cnt);
  hist_kernel<<<(L2 + 255) / 256, 256, 0, stream>>>(idx2, L2, cnt);

  // 3) q = ns @ Wq
  q_kernel<<<(N + 31) / 32, 128, 0, stream>>>(ns, b_Wq, qb, N);

  // 4) fused relation message + attention accumulation
  rel_kernel<1, 2><<<(T0 + 31) / 32, 128, 0, stream>>>(ns, idx0, T0, b_r0W1, r0b1, b_r0W2,
                                                       r0b2, b_Wk, b_Wv, qb, cnt, den, num);
  rel_kernel<2, 2><<<(T1 + 31) / 32, 128, 0, stream>>>(ns, idx1, T1, b_r1W1, r1b1, b_r1W2,
                                                       r1b2, b_Wk, b_Wv, qb, cnt, den, num);
  rel_kernel<3, 1><<<(T2 + 15) / 16, 128, 0, stream>>>(ns, idx2, T2, b_r2W1, r2b1, b_r2W2,
                                                       r2b2, b_Wk, b_Wv, qb, cnt, den, num);

  // 5) final update MLP
  out_kernel<<<(N + 31) / 32, 128, 0, stream>>>(ns, num, den, b_uW1, ub1, b_uW2, ub2,
                                                (float*)d_out, N);
}
